// SS2D_22282290331780
// MI455X (gfx1250) — matrix operand order repaired and checked by reading, not compiled, not run on hardware
//
#include <hip/hip_runtime.h>
#include <hip/hip_bf16.h>
#include <cmath>

#define Bb   4
#define Hh   48
#define Ww   48
#define DM   96
#define DI   192
#define Nn   16
#define Rr   6
#define Kk   4
#define REDd 24
#define Ll   (Hh*Ww)        // 2304
#define CD   (Rr + 2*Nn)    // 38
#define ML   (Bb*Ll)        // 9216

typedef __attribute__((ext_vector_type(16))) __bf16 v16bf;
typedef __attribute__((ext_vector_type(8)))  float  v8f;

// fast activations: v_rcp_f32 instead of IEEE divide chains
__device__ __forceinline__ float fastrcp(float x){ return __builtin_amdgcn_rcpf(x); }
__device__ __forceinline__ float sigmf(float x){ return fastrcp(1.f + __expf(-x)); }
__device__ __forceinline__ float siluf(float x){ return x * sigmf(x); }
__device__ __forceinline__ float softplusf(float x){ return x > 20.f ? x : log1pf(__expf(x)); }
__device__ __forceinline__ float geluf(float x){ return 0.5f * x * (1.f + erff(x * 0.7071067811865476f)); }

// scan-order position l -> row-major token index (same map serves gather AND reverse-scatter)
__device__ __forceinline__ int scan_idx(int k, int l){
  switch(k){
    case 0:  return l;
    case 1:  return Ll - 1 - l;
    case 2:  return (l % Hh) * Ww + (l / Hh);
    default: { int j = Ll - 1 - l; return (j % Hh) * Ww + (j / Hh); }
  }
}

// ---- WMMA fragment builders (CDNA5 16x16x32 bf16 layouts) ----
// A (16x32): lane holds row m=lane&15; K index = i + 8*(lane>>4) + (i>=8 ? 8 : 0)
template<typename F>
__device__ __forceinline__ v16bf frag_a(F f, int lane){
  v16bf a; int m = lane & 15, g = lane >> 4;
  #pragma unroll
  for (int i = 0; i < 16; ++i){
    int kk = i + 8*g + ((i >= 8) ? 8 : 0);
    a[i] = (__bf16)f(m, kk);
  }
  return a;
}
// B (32x16): lane holds col n=lane&15; K index = i + 8*(lane>>4) + (i>=8 ? 8 : 0)
template<typename F>
__device__ __forceinline__ v16bf frag_b(F f, int lane){
  v16bf b; int n = lane & 15, g = lane >> 4;
  #pragma unroll
  for (int i = 0; i < 16; ++i){
    int kk = i + 8*g + ((i >= 8) ? 8 : 0);
    b[i] = (__bf16)f(kk, n);
  }
  return b;
}

// ---------- K1: in_proj GEMM (M=9216, K=96, N=384), 2 N-tiles per wave, + split + silu(z) ----------
__global__ void k_in_proj(const float* __restrict__ x, const float* __restrict__ w,
                          float* __restrict__ xi, float* __restrict__ z){
  int wave = (blockIdx.x * blockDim.x + threadIdx.x) >> 5;
  int lane = threadIdx.x & 31;
  const int NP = (2*DI)/32;                 // 12 column-tile PAIRS
  int mt = wave / NP, np = wave % NP;
  if (mt >= ML/16) return;
  int m0 = mt*16, n0 = np*32;
  v8f c0 = {}, c1 = {};
  #pragma unroll
  for (int k0 = 0; k0 < DM; k0 += 32){
    v16bf a  = frag_a([&](int ml,int kl){ return x[(m0+ml)*DM + k0 + kl]; }, lane);
    v16bf b0 = frag_b([&](int kl,int nl){ return w[(n0+nl)*DM + k0 + kl]; }, lane);
    v16bf b1 = frag_b([&](int kl,int nl){ return w[(n0+16+nl)*DM + k0 + kl]; }, lane);
    c0 = __builtin_amdgcn_wmma_f32_16x16x32_bf16(false, a, false, b0, (short)0, c0, false, false);
    c1 = __builtin_amdgcn_wmma_f32_16x16x32_bf16(false, a, false, b1, (short)0, c1, false, false);
  }
  int nl = lane & 15, g = lane >> 4;
  #pragma unroll
  for (int j = 0; j < 8; ++j){
    int m = m0 + j + 8*g;
    int na = n0 + nl, nb2 = n0 + 16 + nl;
    float va = c0[j], vb = c1[j];
    if (na  < DI) xi[m*DI + na]        = va; else z[m*DI + (na  - DI)] = siluf(va);
    if (nb2 < DI) xi[m*DI + nb2]       = vb; else z[m*DI + (nb2 - DI)] = siluf(vb);
  }
}

// ---------- K2: depthwise 3x3 conv + bias + silu ; xc stored (B,DI,L) ----------
__global__ void k_conv(const float* __restrict__ xi, const float* __restrict__ cw,
                       const float* __restrict__ cb, float* __restrict__ xc){
  int idx = blockIdx.x * blockDim.x + threadIdx.x;
  if (idx >= Bb*DI*Ll) return;
  int lp = idx % Ll; int d = (idx / Ll) % DI; int b = idx / (Ll*DI);
  int h = lp / Ww, wq = lp % Ww;
  float acc = cb[d];
  #pragma unroll
  for (int dy = 0; dy < 3; ++dy){
    int hh = h + dy - 1; if (hh < 0 || hh >= Hh) continue;
    #pragma unroll
    for (int dx = 0; dx < 3; ++dx){
      int wv = wq + dx - 1; if (wv < 0 || wv >= Ww) continue;
      acc += xi[(b*Ll + hh*Ww + wv)*DI + d] * cw[d*9 + dy*3 + dx];
    }
  }
  xc[(b*DI + d)*Ll + lp] = siluf(acc);
}

// ---------- K3: x_proj GEMM per (b,k): (38 x 192) @ (192 x 2304) with scan gather ----------
__global__ void k_x_proj(const float* __restrict__ xc, const float* __restrict__ xpw,
                         float* __restrict__ xdbl){
  int wave = (blockIdx.x * blockDim.x + threadIdx.x) >> 5;
  int lane = threadIdx.x & 31;
  const int NT = Ll/16;                       // 144
  const int MT = 3;                           // ceil(38/16)
  int t = wave;
  int nt = t % NT; t /= NT;
  int mt = t % MT; t /= MT;
  int k  = t % Kk; int b = t / Kk;
  if (b >= Bb) return;
  int m0 = mt*16, l0 = nt*16;
  const float* A   = xpw + k*CD*DI;
  const float* xcb = xc  + b*DI*Ll;
  int cidx = scan_idx(k, l0 + (lane & 15));   // per-lane gather column
  v8f c = {};
  #pragma unroll
  for (int k0 = 0; k0 < DI; k0 += 32){
    v16bf a  = frag_a([&](int ml,int kl){ int r = m0+ml; return (r < CD) ? A[r*DI + k0 + kl] : 0.f; }, lane);
    v16bf bb = frag_b([&](int kl,int nl){ (void)nl; return xcb[(k0+kl)*Ll + cidx]; }, lane);
    c = __builtin_amdgcn_wmma_f32_16x16x32_bf16(false, a, false, bb, (short)0, c, false, false);
  }
  int g = lane >> 4, l = l0 + (lane & 15);
  float* outb = xdbl + (long)(b*Kk + k)*CD*Ll;
  #pragma unroll
  for (int j = 0; j < 8; ++j){
    int row = m0 + j + 8*g;
    if (row < CD) outb[row*Ll + l] = c[j];
  }
}

// ---------- K4: fused dt-proj + softplus + selective scan + D*u + reverse-scatter ----------
// lane = state n (16 per row), 2 rows (b,k,d) per wave32; y-reduction via shfl_xor in 16-lane groups
__global__ void k_scan(const float* __restrict__ xc, const float* __restrict__ xdbl,
                       const float* __restrict__ dtw, const float* __restrict__ dtb,
                       const float* __restrict__ A_logs, const float* __restrict__ Ds,
                       float* __restrict__ yst){
  int gid = blockIdx.x * blockDim.x + threadIdx.x;
  int row = gid >> 4;
  int n   = gid & 15;
  if (row >= Bb*Kk*DI) return;
  int d = row % DI; int k = (row / DI) % Kk; int b = row / (DI*Kk);
  int kd = k*DI + d;
  float An   = -__expf(A_logs[kd*Nn + n]);
  float Dv   = Ds[kd];
  float bias = dtb[kd];
  float w0 = dtw[kd*Rr+0], w1 = dtw[kd*Rr+1], w2 = dtw[kd*Rr+2];
  float w3 = dtw[kd*Rr+3], w4 = dtw[kd*Rr+4], w5 = dtw[kd*Rr+5];
  const float* xcb = xc + (b*DI + d)*Ll;
  const float* xd  = xdbl + (long)(b*Kk + k)*CD*Ll;
  const float* Bp  = xd + (Rr + n)*Ll;
  const float* Cp  = xd + (Rr + Nn + n)*Ll;
  float* yo = yst + (long)(b*Kk + k)*Ll*DI + d;
  float h = 0.f;
  for (int l = 0; l < Ll; ++l){
    if ((l & 63) == 0 && l + 512 < Ll){       // CDNA5 prefetch: pull rows 512 elems ahead
      __builtin_prefetch(xd + l + 512, 0, 1);
      __builtin_prefetch(Bp + l + 512, 0, 1);
      __builtin_prefetch(Cp + l + 512, 0, 1);
    }
    int si  = scan_idx(k, l);
    float u = xcb[si];
    float dts = bias + xd[l]*w0 + xd[Ll+l]*w1 + xd[2*Ll+l]*w2
                     + xd[3*Ll+l]*w3 + xd[4*Ll+l]*w4 + xd[5*Ll+l]*w5;
    float dlt = softplusf(dts);
    h = h * __expf(dlt * An) + (dlt * u) * Bp[l];
    float t = h * Cp[l];
    t += __shfl_xor(t, 1, 32); t += __shfl_xor(t, 2, 32);
    t += __shfl_xor(t, 4, 32); t += __shfl_xor(t, 8, 32);
    if (n == 0) yo[(long)si*DI] = t + u*Dv;   // fused multi_reverse + transpose
  }
}

// ---------- K5: per-token LN stats of yst (B,K,L,DI) for the BiAttn gate ----------
__global__ void k_ln_stats(const float* __restrict__ yst, float* __restrict__ mu_, float* __restrict__ rs_){
  int wave = (blockIdx.x * blockDim.x + threadIdx.x) >> 5;
  int lane = threadIdx.x & 31;
  if (wave >= Bb*Kk*Ll) return;
  const float* p = yst + (long)wave*DI;
  float s = 0.f, q = 0.f;
  #pragma unroll
  for (int i = 0; i < DI/32; ++i){ float v = p[lane + 32*i]; s += v; q += v*v; }
  #pragma unroll
  for (int off = 16; off > 0; off >>= 1){ s += __shfl_xor(s, off, 32); q += __shfl_xor(q, off, 32); }
  if (lane == 0){
    float mu = s / DI;
    mu_[wave] = mu;
    rs_[wave] = rsqrtf(q/DI - mu*mu + 1e-5f);
  }
}

// ---------- K6: deterministic gate mean over L: g[b,k,d] = nw[d]*mean_l((x-mu)*rs) + nb[d] ----------
__global__ void k_gate_mean(const float* __restrict__ yst, const float* __restrict__ mu_,
                            const float* __restrict__ rs_, const float* __restrict__ nw,
                            const float* __restrict__ nb, float* __restrict__ g){
  int idx = blockIdx.x * blockDim.x + threadIdx.x;
  if (idx >= Bb*Kk*DI) return;
  int d = idx % DI; int bk = idx / DI;
  const float* p  = yst + (long)bk*Ll*DI + d;
  const float* mm = mu_ + bk*Ll;
  const float* rr = rs_ + bk*Ll;
  float acc = 0.f;
  for (int l = 0; l < Ll; ++l) acc += (p[(long)l*DI] - mm[l]) * rr[l];
  g[idx] = (acc / (float)Ll) * nw[d] + nb[d];
}

// ---------- K7: gate MLP (gelu + sigmoid), one block per (b,k) ----------
__global__ void k_gate_mlp(const float* __restrict__ g, const float* __restrict__ rw,
                           const float* __restrict__ rb, const float* __restrict__ sw,
                           const float* __restrict__ sb, float* __restrict__ cg){
  __shared__ float g1[REDd];
  int bk = blockIdx.x, t = threadIdx.x;
  const float* gg = g + bk*DI;
  if (t < REDd){
    float acc = rb[t];
    for (int d2 = 0; d2 < DI; ++d2) acc += gg[d2] * rw[t*DI + d2];
    g1[t] = geluf(acc);
  }
  __syncthreads();
  if (t < DI){
    float acc = sb[t];
    #pragma unroll
    for (int r = 0; r < REDd; ++r) acc += g1[r] * sw[t*REDd + r];
    cg[bk*DI + t] = sigmf(acc);
  }
}

// ---------- K8: gate*sum_k + out LayerNorm + multiply by silu(z) ----------
__global__ void k_combine(const float* __restrict__ yst, const float* __restrict__ cg,
                          const float* __restrict__ onw, const float* __restrict__ onb,
                          const float* __restrict__ z, float* __restrict__ yz){
  int wave = (blockIdx.x * blockDim.x + threadIdx.x) >> 5;
  int lane = threadIdx.x & 31;
  if (wave >= Bb*Ll) return;
  int b = wave / Ll, l = wave % Ll;
  float v[DI/32];
  float s = 0.f, q = 0.f;
  #pragma unroll
  for (int i = 0; i < DI/32; ++i){
    int d = lane + 32*i;
    float acc = 0.f;
    #pragma unroll
    for (int k = 0; k < Kk; ++k){
      int bk = b*Kk + k;
      acc += yst[((long)bk*Ll + l)*DI + d] * cg[bk*DI + d];
    }
    v[i] = acc; s += acc; q += acc*acc;
  }
  #pragma unroll
  for (int off = 16; off > 0; off >>= 1){ s += __shfl_xor(s, off, 32); q += __shfl_xor(q, off, 32); }
  float mu  = s / DI;
  float rsv = rsqrtf(q/DI - mu*mu + 1e-5f);
  #pragma unroll
  for (int i = 0; i < DI/32; ++i){
    int d = lane + 32*i;
    float yn = (v[i] - mu) * rsv * onw[d] + onb[d];
    yz[(long)wave*DI + d] = yn * z[(long)wave*DI + d];
  }
}

// ---------- K9: out_proj GEMM (M=9216, K=192, N=96), 2 N-tiles per wave ----------
__global__ void k_out_proj(const float* __restrict__ yz, const float* __restrict__ w,
                           float* __restrict__ out){
  int wave = (blockIdx.x * blockDim.x + threadIdx.x) >> 5;
  int lane = threadIdx.x & 31;
  const int NP = DM/32;                     // 3 column-tile pairs
  int mt = wave / NP, np = wave % NP;
  if (mt >= ML/16) return;
  int m0 = mt*16, n0 = np*32;
  v8f c0 = {}, c1 = {};
  #pragma unroll
  for (int k0 = 0; k0 < DI; k0 += 32){
    v16bf a  = frag_a([&](int ml,int kl){ return yz[(long)(m0+ml)*DI + k0 + kl]; }, lane);
    v16bf b0 = frag_b([&](int kl,int nl){ return w[(n0+nl)*DI + k0 + kl]; }, lane);
    v16bf b1 = frag_b([&](int kl,int nl){ return w[(n0+16+nl)*DI + k0 + kl]; }, lane);
    c0 = __builtin_amdgcn_wmma_f32_16x16x32_bf16(false, a, false, b0, (short)0, c0, false, false);
    c1 = __builtin_amdgcn_wmma_f32_16x16x32_bf16(false, a, false, b1, (short)0, c1, false, false);
  }
  int nl = lane & 15, g = lane >> 4;
  #pragma unroll
  for (int j = 0; j < 8; ++j){
    int m = m0 + j + 8*g;
    out[(long)m*DM + n0 + nl]      = c0[j];
    out[(long)m*DM + n0 + 16 + nl] = c1[j];
  }
}

extern "C" void kernel_launch(void* const* d_in, const int* in_sizes, int n_in,
                              void* d_out, int out_size, void* d_ws, size_t ws_size,
                              hipStream_t stream) {
  const float* x    = (const float*)d_in[0];
  const float* ipw  = (const float*)d_in[1];
  const float* cw   = (const float*)d_in[2];
  const float* cb   = (const float*)d_in[3];
  const float* xpw  = (const float*)d_in[4];
  const float* dtw  = (const float*)d_in[5];
  const float* dtb  = (const float*)d_in[6];
  const float* Alog = (const float*)d_in[7];
  const float* Dsp  = (const float*)d_in[8];
  const float* anw  = (const float*)d_in[9];
  const float* anb  = (const float*)d_in[10];
  const float* arw  = (const float*)d_in[11];
  const float* arb  = (const float*)d_in[12];
  const float* asw  = (const float*)d_in[13];
  const float* asb  = (const float*)d_in[14];
  const float* onw  = (const float*)d_in[15];
  const float* onb  = (const float*)d_in[16];
  const float* opw  = (const float*)d_in[17];
  float* out = (float*)d_out;

  float* ws = (float*)d_ws;
  const size_t SZ_BLD = (size_t)ML*DI;            // 1,769,472
  float* xi   = ws;                               size_t off = SZ_BLD;
  float* z    = ws + off;                         off += SZ_BLD;
  float* xc   = ws + off;                         off += SZ_BLD;
  float* xdbl = ws + off;                         off += (size_t)Bb*Kk*CD*Ll;   // 1,400,832
  float* yst  = ws + off;                         off += (size_t)Bb*Kk*Ll*DI;   // 7,077,888
  float* mu_  = ws + off;                         off += (size_t)Bb*Kk*Ll;
  float* rs_  = ws + off;                         off += (size_t)Bb*Kk*Ll;
  float* gmu  = ws + off;                         off += (size_t)Bb*Kk*DI;
  float* cg   = ws + off;                         off += (size_t)Bb*Kk*DI;
  float* yz   = ws + off;                         off += SZ_BLD;
  (void)ws_size; (void)n_in; (void)in_sizes; (void)out_size;

  // K1: in_proj — 576 * 12 = 6912 waves (2 N-tiles each), 4 waves/block
  k_in_proj<<<6912/4, 128, 0, stream>>>(x, ipw, xi, z);
  // K2: depthwise conv — 1,769,472 threads
  k_conv<<<(Bb*DI*Ll + 255)/256, 256, 0, stream>>>(xi, cw, cb, xc);
  // K3: x_proj — 4*4*3*144 = 6912 waves
  k_x_proj<<<6912/4, 128, 0, stream>>>(xc, xpw, xdbl);
  // K4: fused dt-proj + selective scan — 49152 lanes
  k_scan<<<(Bb*Kk*DI*16)/128, 128, 0, stream>>>(xc, xdbl, dtw, dtb, Alog, Dsp, yst);
  // K5: per-token LN stats — 36864 waves
  k_ln_stats<<<(Bb*Kk*Ll)/4, 128, 0, stream>>>(yst, mu_, rs_);
  // K6: gate mean — 3072 threads
  k_gate_mean<<<(Bb*Kk*DI + 255)/256, 256, 0, stream>>>(yst, mu_, rs_, anw, anb, gmu);
  // K7: gate MLP — one block per (b,k)
  k_gate_mlp<<<Bb*Kk, DI, 0, stream>>>(gmu, arw, arb, asw, asb, cg);
  // K8: combine + out-LN + z-gate — 9216 waves
  k_combine<<<(Bb*Ll)/4, 128, 0, stream>>>(yst, cg, onw, onb, z, yz);
  // K9: out_proj — 576*3 = 1728 waves
  k_out_proj<<<1728/4, 128, 0, stream>>>(yz, opw, out);
}
